// HNO_80453327389401
// MI455X (gfx1250) — compile-verified, matrix operand order checked by reading, not run on hardware
//
#include <hip/hip_runtime.h>

#define N_NODES 100000
#define N_EDGES 1600000
#define HID 128
#define NF ((size_t)N_NODES * HID)

typedef __bf16 bf16_t;
typedef __attribute__((ext_vector_type(16))) __bf16 v16bf;
typedef __attribute__((ext_vector_type(8)))  __bf16 v8bf;
typedef __attribute__((ext_vector_type(8)))  float  v8f;
typedef __attribute__((ext_vector_type(4)))  float  v4f;

// ---------------- utility kernels ----------------

__global__ void k_zero(float* __restrict__ p, size_t n) {
    size_t i = (size_t)blockIdx.x * blockDim.x + threadIdx.x;
    size_t st = (size_t)gridDim.x * blockDim.x;
    for (; i < n; i += st) p[i] = 0.0f;
}

// deg[src] += (src != dst)
__global__ void k_degree(const int* __restrict__ src, const int* __restrict__ dst,
                         float* __restrict__ deg, int ne) {
    int e = blockIdx.x * blockDim.x + threadIdx.x;
    if (e >= ne) return;
    int s = src[e], d = dst[e];
    if (s != d) atomicAdd(&deg[s], 1.0f);
}

// wn = -dinv[src]*dinv[dst]*(src!=dst)
__global__ void k_wnorm(const int* __restrict__ src, const int* __restrict__ dst,
                        const float* __restrict__ deg, float* __restrict__ wn, int ne) {
    int e = blockIdx.x * blockDim.x + threadIdx.x;
    if (e >= ne) return;
    int s = src[e], d = dst[e];
    float w = 0.0f;
    if (s != d) {
        float ds = deg[s], dd = deg[d];
        float is = ds > 0.0f ? rsqrtf(ds) : 0.0f;
        float id = dd > 0.0f ? rsqrtf(dd) : 0.0f;
        w = -is * id;
    }
    wn[e] = w;
}

// out[dst] += h[src] * wn, width 3
__global__ void k_prop3(const float* __restrict__ h, const float* __restrict__ wn,
                        const int* __restrict__ src, const int* __restrict__ dst,
                        float* __restrict__ out, int ne) {
    int e = blockIdx.x * blockDim.x + threadIdx.x;
    if (e >= ne) return;
    float w = wn[e];
    if (w == 0.0f) return;
    int s = src[e], d = dst[e];
    atomicAdd(&out[(size_t)d * 3 + 0], h[(size_t)s * 3 + 0] * w);
    atomicAdd(&out[(size_t)d * 3 + 1], h[(size_t)s * 3 + 1] * w);
    atomicAdd(&out[(size_t)d * 3 + 2], h[(size_t)s * 3 + 2] * w);
}

// out[dst] += h[src] * wn, width 128; one wave per edge, float4 per lane
__global__ __launch_bounds__(256)
void k_prop128(const float* __restrict__ h, const float* __restrict__ wn,
               const int* __restrict__ src, const int* __restrict__ dst,
               float* __restrict__ out, int ne) {
    int e = blockIdx.x * (blockDim.x >> 5) + (threadIdx.x >> 5);
    if (e >= ne) return;
    float w = wn[e];
    if (w == 0.0f) return;
    int lane = threadIdx.x & 31;
    int s = src[e], d = dst[e];
    v4f v = *(const v4f*)(h + (size_t)s * HID + lane * 4);
    float* o = out + (size_t)d * HID + lane * 4;
    atomicAdd(o + 0, v.x * w);
    atomicAdd(o + 1, v.y * w);
    atomicAdd(o + 2, v.z * w);
    atomicAdd(o + 3, v.w * w);
}

// y = 2*y - prev  (Chebyshev recurrence)
__global__ void k_cheb_combine(float* __restrict__ y, const float* __restrict__ prev, size_t n) {
    size_t i = (size_t)blockIdx.x * blockDim.x + threadIdx.x;
    size_t st = (size_t)gridDim.x * blockDim.x;
    for (; i < n; i += st) y[i] = 2.0f * y[i] - prev[i];
}

// layer-1 combine (input width 3): out[r,c] = sum_k sum_j Tk[r,j]*W[k,j,c]
__global__ void k_gemm_in3(const float* __restrict__ T0, const float* __restrict__ T1,
                           const float* __restrict__ T2, const float* __restrict__ T3,
                           const float* __restrict__ W, float* __restrict__ out, int nrows) {
    int id = blockIdx.x * blockDim.x + threadIdx.x;
    if (id >= nrows * HID) return;
    int row = id >> 7, col = id & 127;
    const float* Ts[4] = {T0, T1, T2, T3};
    float acc = 0.0f;
#pragma unroll
    for (int k = 0; k < 4; ++k) {
        const float* t = Ts[k] + (size_t)row * 3;
        const float* w = W + (size_t)(k * 3) * HID + col;
        acc += t[0] * w[0] + t[1] * w[HID] + t[2] * w[2 * HID];
    }
    out[id] = acc;
}

// transpose+convert weights: Wt[n*512 + k] = (bf16) W[k*128 + n], k in [0,512)
__global__ void k_wconv(const float* __restrict__ W, bf16_t* __restrict__ Wt) {
    int id = blockIdx.x * blockDim.x + threadIdx.x;
    if (id >= HID * 512) return;
    int n = id >> 9, k = id & 511;
    Wt[(size_t)n * 512 + k] = (bf16_t)W[(size_t)k * HID + n];
}

// ------------- WMMA GEMM: out[N x 128] = [T0|T1|T2|T3](N x 512) @ Wt^T -------------
// One wave computes a 16-row x 128-col strip. bf16 multiplies, f32 accumulate.
__global__ __launch_bounds__(256)
void k_gemm512_wmma(const float* __restrict__ T0, const float* __restrict__ T1,
                    const float* __restrict__ T2, const float* __restrict__ T3,
                    const bf16_t* __restrict__ Wt, const float* __restrict__ bias,
                    int addBias, float* __restrict__ out, int nrows) {
    const int lane = threadIdx.x & 31;
    const int wave = threadIdx.x >> 5;
    const int mbase = (blockIdx.x * 8 + wave) * 16;
    if (mbase >= nrows) return;
    const int kh  = lane >> 4;   // half-wave select
    const int l15 = lane & 15;
    int row = mbase + l15;
    if (row >= nrows) row = nrows - 1;   // clamp loads; stores guarded

    v8f acc[8];
#pragma unroll
    for (int i = 0; i < 8; ++i)
#pragma unroll
        for (int j = 0; j < 8; ++j) acc[i][j] = 0.0f;

#pragma unroll
    for (int tk = 0; tk < 4; ++tk) {
        const float* A = (tk == 0) ? T0 : (tk == 1) ? T1 : (tk == 2) ? T2 : T3;
        const float* arowbase = A + (size_t)row * HID;
#pragma unroll
        for (int q = 0; q < 4; ++q) {
            const int kofs = q * 32;
            const int kglob = tk * 128 + kofs;
            // A fragment: 16-bit A layout — lane=M, half selects K groups
            const float* ar = arowbase + kofs + kh * 8;
            v16bf a;
#pragma unroll
            for (int j = 0; j < 8; ++j) a[j]     = (bf16_t)ar[j];
#pragma unroll
            for (int j = 0; j < 8; ++j) a[8 + j] = (bf16_t)ar[16 + j];
#pragma unroll
            for (int nt = 0; nt < 8; ++nt) {
                const bf16_t* wc = Wt + (size_t)(nt * 16 + l15) * 512 + kglob + kh * 8;
                v8bf blo = *(const v8bf*)(wc);
                v8bf bhi = *(const v8bf*)(wc + 16);
                v16bf b = __builtin_shufflevector(blo, bhi,
                    0, 1, 2, 3, 4, 5, 6, 7, 8, 9, 10, 11, 12, 13, 14, 15);
                acc[nt] = __builtin_amdgcn_wmma_f32_16x16x32_bf16(
                    false, a, false, b, (short)0, acc[nt], false, false);
            }
        }
    }

    // D layout: VGPR r, lane L -> M = r + 8*(L>=16), N = L%16
#pragma unroll
    for (int nt = 0; nt < 8; ++nt) {
        int col = nt * 16 + l15;
        float bv = addBias ? bias[col] : 0.0f;
#pragma unroll
        for (int r = 0; r < 8; ++r) {
            int m = mbase + r + 8 * kh;
            if (m < nrows) out[(size_t)m * HID + col] = acc[nt][r] + bv;
        }
    }
}

// ---------------- BatchNorm helpers ----------------

#define ROWS_PER_BLK 512
__global__ __launch_bounds__(128)
void k_colstats(const float* __restrict__ x, float* __restrict__ s1,
                float* __restrict__ s2, int nrows) {
    int col = threadIdx.x;
    int r0 = blockIdx.x * ROWS_PER_BLK;
    int r1 = min(r0 + ROWS_PER_BLK, nrows);
    float a = 0.0f, b = 0.0f;
    for (int r = r0; r < r1; ++r) {
        float v = x[(size_t)r * HID + col];
        a += v; b += v * v;
    }
    atomicAdd(&s1[col], a);
    atomicAdd(&s2[col], b);
}

__global__ void k_bnfinal(const float* __restrict__ s1, const float* __restrict__ s2,
                          const float* __restrict__ g, const float* __restrict__ be,
                          float* __restrict__ scale, float* __restrict__ shift, float invN) {
    int c = threadIdx.x;
    float mu = s1[c] * invN;
    float var = fmaxf(s2[c] * invN - mu * mu, 0.0f);
    float sc = g[c] * rsqrtf(var + 1e-5f);
    scale[c] = sc;
    shift[c] = be[c] - mu * sc;
}

// y = act(x*scale[col] + shift[col]); slope=0.01 => leaky_relu, 0 => relu
__global__ void k_bnact(const float* __restrict__ x, const float* __restrict__ scale,
                        const float* __restrict__ shift, float* __restrict__ y,
                        size_t n, float slope) {
    size_t i = (size_t)blockIdx.x * blockDim.x + threadIdx.x;
    size_t st = (size_t)gridDim.x * blockDim.x;
    for (; i < n; i += st) {
        int col = (int)(i & 127);
        float v = x[i] * scale[col] + shift[col];
        y[i] = v > 0.0f ? v : v * slope;
    }
}

// ---------------- L2-normalize + 128->3 head ----------------
__global__ __launch_bounds__(256)
void k_l2head(const float* __restrict__ h, const float* __restrict__ Wm,
              const float* __restrict__ bm, float* __restrict__ out, int nrows) {
    int row = blockIdx.x * (blockDim.x >> 5) + (threadIdx.x >> 5);
    if (row >= nrows) return;
    int lane = threadIdx.x & 31;
    v4f v = *(const v4f*)(h + (size_t)row * HID + lane * 4);
    float ss = v.x * v.x + v.y * v.y + v.z * v.z + v.w * v.w;
#pragma unroll
    for (int o = 16; o > 0; o >>= 1) ss += __shfl_xor(ss, o, 32);
    float inv = 1.0f / fmaxf(sqrtf(ss), 1e-12f);
    int j = lane * 4;
#pragma unroll
    for (int c = 0; c < 3; ++c) {
        float p = v.x * Wm[(j + 0) * 3 + c] + v.y * Wm[(j + 1) * 3 + c]
                + v.z * Wm[(j + 2) * 3 + c] + v.w * Wm[(j + 3) * 3 + c];
#pragma unroll
        for (int o = 16; o > 0; o >>= 1) p += __shfl_xor(p, o, 32);
        if (lane == 0) out[(size_t)row * 3 + c] = p * inv + bm[c];
    }
}

// ---------------- host orchestration ----------------

static void cheb_props128(float* T0, float* T1, float* T2, float* T3,
                          const float* wn, const int* src, const int* dst,
                          hipStream_t stream) {
    int pblk = (N_EDGES + 7) / 8;
    k_zero<<<4096, 256, 0, stream>>>(T1, NF);
    k_prop128<<<pblk, 256, 0, stream>>>(T0, wn, src, dst, T1, N_EDGES);
    k_zero<<<4096, 256, 0, stream>>>(T2, NF);
    k_prop128<<<pblk, 256, 0, stream>>>(T1, wn, src, dst, T2, N_EDGES);
    k_cheb_combine<<<4096, 256, 0, stream>>>(T2, T0, NF);
    k_zero<<<4096, 256, 0, stream>>>(T3, NF);
    k_prop128<<<pblk, 256, 0, stream>>>(T2, wn, src, dst, T3, N_EDGES);
    k_cheb_combine<<<4096, 256, 0, stream>>>(T3, T1, NF);
}

extern "C" void kernel_launch(void* const* d_in, const int* in_sizes, int n_in,
                              void* d_out, int out_size, void* d_ws, size_t ws_size,
                              hipStream_t stream) {
    const float* x   = (const float*)d_in[0];
    const int*   ei  = (const int*)d_in[1];
    const int*   src = ei;
    const int*   dst = ei + N_EDGES;
    const float* W1  = (const float*)d_in[2];
    const float* g1  = (const float*)d_in[4];
    const float* be1 = (const float*)d_in[5];
    const float* W2  = (const float*)d_in[6];
    const float* g2  = (const float*)d_in[8];
    const float* be2 = (const float*)d_in[9];
    const float* W3  = (const float*)d_in[10];
    const float* g3  = (const float*)d_in[12];
    const float* be3 = (const float*)d_in[13];
    const float* W4  = (const float*)d_in[14];
    const float* b4  = (const float*)d_in[15];
    const float* Wm  = (const float*)d_in[16];
    const float* bm  = (const float*)d_in[17];
    (void)in_sizes; (void)n_in; (void)out_size; (void)ws_size;

    // carve workspace (floats, 256B aligned chunks)
    float* p = (float*)d_ws;
    auto take = [&](size_t n) { float* r = p; p += ((n + 63) & ~(size_t)63); return r; };
    float*  deg = take(N_NODES);
    float*  wn  = take(N_EDGES);
    float*  s1  = take(HID);
    float*  s2  = take(HID);
    float*  sc  = take(HID);
    float*  sh  = take(HID);
    bf16_t* Wt  = (bf16_t*)take(HID * 512 / 2);   // 128x512 bf16
    float*  T0  = take(NF);
    float*  T1  = take(NF);
    float*  T2  = take(NF);
    float*  T3  = take(NF);
    float*  OUT = take(NF);
    float*  u1  = take((size_t)N_NODES * 3);
    float*  u2  = take((size_t)N_NODES * 3);
    float*  u3  = take((size_t)N_NODES * 3);

    const int eb = (N_EDGES + 255) / 256;
    const int gemmBlocks = (N_NODES + 127) / 128;
    const int statBlocks = (N_NODES + ROWS_PER_BLK - 1) / ROWS_PER_BLK;
    const float invN = 1.0f / (float)N_NODES;

    // edge norms
    k_zero<<<512, 256, 0, stream>>>(deg, N_NODES);
    k_degree<<<eb, 256, 0, stream>>>(src, dst, deg, N_EDGES);
    k_wnorm<<<eb, 256, 0, stream>>>(src, dst, deg, wn, N_EDGES);

    // ---- layer 1 (input width 3) ----
    k_zero<<<2048, 256, 0, stream>>>(u1, (size_t)N_NODES * 3);
    k_prop3<<<eb, 256, 0, stream>>>(x, wn, src, dst, u1, N_EDGES);
    k_zero<<<2048, 256, 0, stream>>>(u2, (size_t)N_NODES * 3);
    k_prop3<<<eb, 256, 0, stream>>>(u1, wn, src, dst, u2, N_EDGES);
    k_cheb_combine<<<2048, 256, 0, stream>>>(u2, x, (size_t)N_NODES * 3);
    k_zero<<<2048, 256, 0, stream>>>(u3, (size_t)N_NODES * 3);
    k_prop3<<<eb, 256, 0, stream>>>(u2, wn, src, dst, u3, N_EDGES);
    k_cheb_combine<<<2048, 256, 0, stream>>>(u3, u1, (size_t)N_NODES * 3);
    k_gemm_in3<<<(N_NODES * HID + 255) / 256, 256, 0, stream>>>(x, u1, u2, u3, W1, OUT, N_NODES);
    k_zero<<<1, 256, 0, stream>>>(s1, HID);
    k_zero<<<1, 256, 0, stream>>>(s2, HID);
    k_colstats<<<statBlocks, 128, 0, stream>>>(OUT, s1, s2, N_NODES);
    k_bnfinal<<<1, 128, 0, stream>>>(s1, s2, g1, be1, sc, sh, invN);
    k_bnact<<<4096, 256, 0, stream>>>(OUT, sc, sh, T0, NF, 0.01f);   // leaky_relu

    // ---- layers 2 & 3 (WMMA combine) ----
    const float* Ws[2]  = {W2, W3};
    const float* gs[2]  = {g2, g3};
    const float* bes[2] = {be2, be3};
    const float slopes[2] = {0.01f, 0.0f};   // leaky, relu
    for (int L = 0; L < 2; ++L) {
        cheb_props128(T0, T1, T2, T3, wn, src, dst, stream);
        k_wconv<<<(HID * 512 + 255) / 256, 256, 0, stream>>>(Ws[L], Wt);
        k_gemm512_wmma<<<gemmBlocks, 256, 0, stream>>>(T0, T1, T2, T3, Wt, nullptr, 0, OUT, N_NODES);
        k_zero<<<1, 256, 0, stream>>>(s1, HID);
        k_zero<<<1, 256, 0, stream>>>(s2, HID);
        k_colstats<<<statBlocks, 128, 0, stream>>>(OUT, s1, s2, N_NODES);
        k_bnfinal<<<1, 128, 0, stream>>>(s1, s2, gs[L], bes[L], sc, sh, invN);
        k_bnact<<<4096, 256, 0, stream>>>(OUT, sc, sh, T0, NF, slopes[L]);
    }

    // ---- layer 4 (WMMA combine + bias, no BN) ----
    cheb_props128(T0, T1, T2, T3, wn, src, dst, stream);
    k_wconv<<<(HID * 512 + 255) / 256, 256, 0, stream>>>(W4, Wt);
    k_gemm512_wmma<<<gemmBlocks, 256, 0, stream>>>(T0, T1, T2, T3, Wt, b4, 1, OUT, N_NODES);

    // ---- L2 normalize + linear head ----
    k_l2head<<<(N_NODES + 7) / 8, 256, 0, stream>>>(OUT, Wm, bm, (float*)d_out, N_NODES);
}
